// Feature_encoder_decoder_27943057228466
// MI455X (gfx1250) — compile-verified, matrix operand order checked by reading, not run on hardware
//
#include <hip/hip_runtime.h>
#include <hip/hip_bf16.h>
#include <cmath>

typedef __attribute__((ext_vector_type(16))) _Float16 v16h;
typedef __attribute__((ext_vector_type(8)))  _Float16 v8h;
typedef __attribute__((ext_vector_type(8)))  float    v8f;
typedef __attribute__((ext_vector_type(4)))  float    v4f;

#define BB 32      // batch
#define SS 128     // seq
#define VV 512     // vocab / inv_size
#define FF 64      // feature dim
#define HH 256     // hidden dim
#define NSTEP (BB * SS)   // 4096 decoder steps

// ---------------------------------------------------------------------------
// WMMA helpers (f16 inputs, f32 accumulate, 16x16x32)
// ---------------------------------------------------------------------------
__device__ __forceinline__ v8f wmma16(v16h a, v16h b, v8f c) {
  // (neg_a, A, neg_b, B, c_mod, C, reuse_a, reuse_b)
  return __builtin_amdgcn_wmma_f32_16x16x32_f16(false, a, false, b, (short)0, c,
                                                false, false);
}

// Load a 16x32 f16 fragment from a row-major matrix (leading dim ld, halfs).
// ISA 7.12.2 layout: lanes 0-15 -> M=lane, K = {g*8+0..7, 16+g*8+0..7}.
// Per lane this is exactly two contiguous 16-byte chunks -> two b128 loads.
// Same routine serves B fragments when weights are stored [N][K] row-major.
__device__ __forceinline__ v16h load_frag16(const _Float16* __restrict__ base,
                                            int ld, int lane) {
  const int g = (lane >> 4) & 1;
  const int r = lane & 15;
  const _Float16* row = base + r * ld + g * 8;
  v8h lo = *(const v8h*)(row);        // K = g*8 + 0..7
  v8h hi = *(const v8h*)(row + 16);   // K = 16 + g*8 + 0..7
  return __builtin_shufflevector(lo, hi, 0, 1, 2, 3, 4, 5, 6, 7,
                                 8, 9, 10, 11, 12, 13, 14, 15);
}

// Same layout, gathered from a per-lane f32 row (feature gather);
// rowbase already points at this lane's M row + k offset. 4x float4 loads.
__device__ __forceinline__ v16h load_frag16_f32row(const float* __restrict__ rowbase,
                                                   int lane) {
  const int g = (lane >> 4) & 1;
  const float* row = rowbase + g * 8;
  v4f f0 = *(const v4f*)(row);
  v4f f1 = *(const v4f*)(row + 4);
  v4f f2 = *(const v4f*)(row + 16);
  v4f f3 = *(const v4f*)(row + 20);
  v16h a;
#pragma unroll
  for (int j = 0; j < 4; ++j) {
    a[j]      = (_Float16)f0[j];
    a[4 + j]  = (_Float16)f1[j];
    a[8 + j]  = (_Float16)f2[j];
    a[12 + j] = (_Float16)f3[j];
  }
  return a;
}

__device__ __forceinline__ float wave_max(float v) {
#pragma unroll
  for (int off = 16; off > 0; off >>= 1) v = fmaxf(v, __shfl_xor(v, off, 32));
  return v;
}
__device__ __forceinline__ float wave_sum(float v) {
#pragma unroll
  for (int off = 16; off > 0; off >>= 1) v += __shfl_xor(v, off, 32);
  return v;
}

// Fast tanh: (e^{2x}-1)/(e^{2x}+1) with clamp so e^{2x} stays finite.
// v_exp_f32 + v_rcp_f32 instead of the OCML tanh call.
__device__ __forceinline__ float tanh_fast(float x) {
  float cx = fminf(fmaxf(x, -10.f), 10.f);
  float e = __expf(2.f * cx);
  return (e - 1.f) * __builtin_amdgcn_rcpf(e + 1.f);
}

// ---------------------------------------------------------------------------
// Kernel 0: f32 -> f16 weight conversion, 8 elements / thread, vector ld/st
// ---------------------------------------------------------------------------
__global__ void cvt_f16_kernel(const float* __restrict__ src,
                               _Float16* __restrict__ dst, int n8) {
  int i = blockIdx.x * blockDim.x + threadIdx.x;
  if (i < n8) {
    v4f f0 = *(const v4f*)(src + (size_t)i * 8);
    v4f f1 = *(const v4f*)(src + (size_t)i * 8 + 4);
    v8h h;
#pragma unroll
    for (int j = 0; j < 4; ++j) {
      h[j]     = (_Float16)f0[j];
      h[4 + j] = (_Float16)f1[j];
    }
    *(v8h*)(dst + (size_t)i * 8) = h;
  }
}

// ---------------------------------------------------------------------------
// Kernel 1: gather + encoder input projection
//   xw[s*32+b][h] = features[batch[b][s]] @ enc_w_ih^T + enc_b_ih   (f32 out)
// 256x16 grid of 16x16 tiles; 8 waves / block, 512 blocks.
// ---------------------------------------------------------------------------
__global__ __launch_bounds__(256) void proj_kernel(
    const int* __restrict__ batch, const float* __restrict__ features,
    const _Float16* __restrict__ w_ih16 /*[HH][FF]*/,
    const float* __restrict__ b_ih, float* __restrict__ xw /*[4096][HH]*/) {
  const int wv = threadIdx.x >> 5;
  const int lane = threadIdx.x & 31;
  const int tile = blockIdx.x * 8 + wv;   // 0..4095
  const int mtile = tile >> 4;            // 0..255
  const int ntile = tile & 15;            // 0..15
  const int g = (lane >> 4) & 1;
  const int r = lane & 15;

  // this lane's A row: global row = s*32+b
  const int rowg = mtile * 16 + r;
  const int b = rowg & 31;
  const int s = rowg >> 5;
  const float* frow = features + (size_t)batch[b * SS + s] * FF;

  v8f acc = {};
#pragma unroll
  for (int kk = 0; kk < 2; ++kk) {
    v16h a = load_frag16_f32row(frow + kk * 32, lane);
    v16h bb = load_frag16(w_ih16 + (ntile * 16) * FF + kk * 32, FF, lane);
    acc = wmma16(a, bb, acc);
  }
  const int col = ntile * 16 + r;
  const float bias = b_ih[col];
#pragma unroll
  for (int i = 0; i < 8; ++i) {
    const int row = mtile * 16 + i + 8 * g;
    xw[row * HH + col] = acc[i] + bias;
  }
}

// ---------------------------------------------------------------------------
// Kernel 2: encoder RNN scan (128 sequential steps, one persistent workgroup)
//   h = tanh(xw[t] + h @ enc_w_hh^T + b_hh)
// ---------------------------------------------------------------------------
__global__ __launch_bounds__(1024) void encoder_kernel(
    const _Float16* __restrict__ w_hh16 /*[HH][HH]*/,
    const float* __restrict__ xw /*[SS*BB][HH]*/,
    const float* __restrict__ b_hh, _Float16* __restrict__ h_out /*[BB*HH]*/) {
  __shared__ _Float16 s_h[BB * HH];   // 16 KB

  const int tid = threadIdx.x;
  const int wv = tid >> 5;
  const int lane = tid & 31;
  const int g = (lane >> 4) & 1;
  const int r = lane & 15;
  const int mt = wv & 1;        // 2 M tiles
  const int nt = wv >> 1;       // 16 N tiles
  const int col = nt * 16 + r;  // this lane's output column (fixed)
  const float bias = b_hh[col]; // loop-invariant, kept in a register

  for (int i = tid; i < BB * HH; i += 1024) s_h[i] = (_Float16)0.f;
  __syncthreads();

  for (int t = 0; t < SS; ++t) {
    v8f acc = {};
    const _Float16* aB = s_h + (mt * 16) * HH;
    const _Float16* bB = w_hh16 + (nt * 16) * HH;
#pragma unroll
    for (int kk = 0; kk < 8; ++kk) {
      v16h a = load_frag16(aB + kk * 32, HH, lane);
      v16h b = load_frag16(bB + kk * 32, HH, lane);
      acc = wmma16(a, b, acc);
    }
    __syncthreads();   // all reads of old s_h done
#pragma unroll
    for (int i = 0; i < 8; ++i) {
      const int row = mt * 16 + i + 8 * g;
      const float v = acc[i] + xw[(t * BB + row) * HH + col] + bias;
      s_h[row * HH + col] = (_Float16)tanh_fast(v);
    }
    __syncthreads();
  }
  for (int i = tid; i < BB * HH; i += 1024) h_out[i] = s_h[i];
}

// ---------------------------------------------------------------------------
// Kernel 3: decoder — 4096 strictly sequential steps on one WGP.
//   h   = tanh(out @ dec_w_ih^T + b_ih + h @ dec_w_hh^T + b_hh)
//   out = softmax(h @ out_w^T + out_b)
// State lives in LDS; f16 weights stream from L2 every step (640 KB/step,
// trivially L2-resident).  s_ov is overlaid: f16 logits (phase 3/4) and f16
// softmax out (phase 4 / next step's A matrix).  The overlay is race-free
// without an extra barrier because wave w reads and writes *only* row w, at
// identical addresses per lane.
// 4 barriers per step.
// ---------------------------------------------------------------------------
__global__ __launch_bounds__(1024) void decoder_kernel(
    const _Float16* __restrict__ w_ih16 /*[HH][VV]*/,
    const _Float16* __restrict__ w_hh16 /*[HH][HH]*/,
    const _Float16* __restrict__ w_out16 /*[VV][HH]*/,
    const float* __restrict__ b_ih, const float* __restrict__ b_hh,
    const float* __restrict__ b_out,
    const _Float16* __restrict__ h_init /*[BB*HH]*/,
    float* __restrict__ out /*[BB][NSTEP][VV]*/) {
  __shared__ _Float16 s_ov[BB * VV];   // 32 KB: out (f16) / logits (f16) overlay
  __shared__ _Float16 s_h[BB * HH];    // 16 KB

  const int tid = threadIdx.x;
  const int wv = tid >> 5;
  const int lane = tid & 31;
  const int g = (lane >> 4) & 1;
  const int r = lane & 15;
  const int mt = wv & 1;        // 2 M tiles of the [32 x 256] hidden update
  const int nt = wv >> 1;       // 16 N tiles
  // phase-3 tile assignment (both tiles share mt2 == mt)
  const int nt2a = wv >> 1;     // 0..15
  const int nt2b = nt2a + 16;   // 16..31

  // loop-invariant per-lane biases -> registers (no LDS bias arrays)
  const int col_h = nt * 16 + r;
  const float bias_h = b_ih[col_h] + b_hh[col_h];
  const int col_a = nt2a * 16 + r;
  const int col_b = nt2b * 16 + r;
  const float ob_a = b_out[col_a];
  const float ob_b = b_out[col_b];

  for (int i = tid; i < BB * VV; i += 1024) s_ov[i] = (_Float16)0.f;  // out0 = 0
  for (int i = tid; i < BB * HH; i += 1024) s_h[i] = h_init[i];
  __syncthreads();

  for (int t = 0; t < NSTEP; ++t) {
    // ---- phase 1: pre_h tile = out@W_ih^T + h@W_hh^T (wave = one 16x16 tile)
    v8f acc = {};
    {
      const _Float16* aB = s_ov + (mt * 16) * VV;
      const _Float16* bB = w_ih16 + (nt * 16) * VV;
#pragma unroll
      for (int kk = 0; kk < 16; ++kk) {   // K = 512
        v16h a = load_frag16(aB + kk * 32, VV, lane);
        v16h b = load_frag16(bB + kk * 32, VV, lane);
        acc = wmma16(a, b, acc);
      }
    }
    {
      const _Float16* aB = s_h + (mt * 16) * HH;
      const _Float16* bB = w_hh16 + (nt * 16) * HH;
#pragma unroll
      for (int kk = 0; kk < 8; ++kk) {    // K = 256
        v16h a = load_frag16(aB + kk * 32, HH, lane);
        v16h b = load_frag16(bB + kk * 32, HH, lane);
        acc = wmma16(a, b, acc);
      }
    }
    __syncthreads();   // (1) all reads of old s_h / s_ov complete

    // ---- phase 2: h = tanh(pre_h + b)
#pragma unroll
    for (int i = 0; i < 8; ++i) {
      const int row = mt * 16 + i + 8 * g;
      s_h[row * HH + col_h] = (_Float16)tanh_fast(acc[i] + bias_h);
    }
    __syncthreads();   // (2) new h visible

    // ---- phase 3: logits = h @ out_w^T + out_b.
    // 64 tiles, 2 per wave; both tiles share the same M tile, so the A
    // fragment is loaded once per k-step and fed to two WMMAs.
    {
      v8f acc2a = {};
      v8f acc2b = {};
      const _Float16* aB = s_h + (mt * 16) * HH;
      const _Float16* b1B = w_out16 + (nt2a * 16) * HH;
      const _Float16* b2B = w_out16 + (nt2b * 16) * HH;
#pragma unroll
      for (int kk = 0; kk < 8; ++kk) {
        v16h a = load_frag16(aB + kk * 32, HH, lane);
        v16h b1 = load_frag16(b1B + kk * 32, HH, lane);
        v16h b2 = load_frag16(b2B + kk * 32, HH, lane);
        acc2a = wmma16(a, b1, acc2a);
        acc2b = wmma16(a, b2, acc2b);
      }
#pragma unroll
      for (int i = 0; i < 8; ++i) {
        const int row = mt * 16 + i + 8 * g;
        s_ov[row * VV + col_a] = (_Float16)(acc2a[i] + ob_a);
        s_ov[row * VV + col_b] = (_Float16)(acc2b[i] + ob_b);
      }
    }
    __syncthreads();   // (3) logits visible

    // ---- phase 4: softmax per row (wave w -> batch row w), write output.
    // Each lane owns 16 contiguous columns of its row: b128 LDS load, b128
    // global stores, b128 LDS writeback.  Lane-private overlay -> no barrier
    // between the read and the writeback.
    {
      const int row = wv;
      const _Float16* lrow = s_ov + row * VV + lane * 16;
      v8h l0 = *(const v8h*)(lrow);
      v8h l1 = *(const v8h*)(lrow + 8);
      float x[16];
#pragma unroll
      for (int j = 0; j < 8; ++j) {
        x[j] = (float)l0[j];
        x[8 + j] = (float)l1[j];
      }
      float m = -1e30f;
#pragma unroll
      for (int j = 0; j < 16; ++j) m = fmaxf(m, x[j]);
      m = wave_max(m);
      float ssum = 0.f;
#pragma unroll
      for (int j = 0; j < 16; ++j) { x[j] = __expf(x[j] - m); ssum += x[j]; }
      ssum = wave_sum(ssum);
      const float inv = __builtin_amdgcn_rcpf(ssum);
      float* orow = out + (size_t)row * ((size_t)NSTEP * VV) + (size_t)t * VV +
                    lane * 16;
      v8h o0, o1;
#pragma unroll
      for (int q = 0; q < 4; ++q) {
        v4f p;
#pragma unroll
        for (int j = 0; j < 4; ++j) p[j] = x[q * 4 + j] * inv;
        *(v4f*)(orow + q * 4) = p;
#pragma unroll
        for (int j = 0; j < 4; ++j) {
          if (q < 2) o0[q * 4 + j] = (_Float16)p[j];
          else       o1[(q - 2) * 4 + j] = (_Float16)p[j];
        }
      }
      _Float16* wrow = s_ov + row * VV + lane * 16;
      *(v8h*)(wrow) = o0;
      *(v8h*)(wrow + 8) = o1;
    }
    __syncthreads();   // (4) new out visible for next step
  }
}

// ---------------------------------------------------------------------------
// Host launcher
// ---------------------------------------------------------------------------
extern "C" void kernel_launch(void* const* d_in, const int* in_sizes, int n_in,
                              void* d_out, int out_size, void* d_ws, size_t ws_size,
                              hipStream_t stream) {
  (void)in_sizes; (void)n_in; (void)out_size; (void)ws_size;

  const int*   batch    = (const int*)d_in[0];
  const float* features = (const float*)d_in[1];
  const float* enc_w_ih = (const float*)d_in[2];
  const float* enc_w_hh = (const float*)d_in[3];
  const float* enc_b_ih = (const float*)d_in[4];
  const float* enc_b_hh = (const float*)d_in[5];
  const float* dec_w_ih = (const float*)d_in[6];
  const float* dec_w_hh = (const float*)d_in[7];
  const float* dec_b_ih = (const float*)d_in[8];
  const float* dec_b_hh = (const float*)d_in[9];
  const float* out_w    = (const float*)d_in[10];
  const float* out_b    = (const float*)d_in[11];
  float* out = (float*)d_out;

  // workspace carve-up
  char* ws = (char*)d_ws;
  float* xw = (float*)ws;              ws += (size_t)SS * BB * HH * sizeof(float);
  _Float16* encwih16 = (_Float16*)ws;  ws += (size_t)HH * FF * sizeof(_Float16);
  _Float16* encwhh16 = (_Float16*)ws;  ws += (size_t)HH * HH * sizeof(_Float16);
  _Float16* decwih16 = (_Float16*)ws;  ws += (size_t)HH * VV * sizeof(_Float16);
  _Float16* decwhh16 = (_Float16*)ws;  ws += (size_t)HH * HH * sizeof(_Float16);
  _Float16* outw16   = (_Float16*)ws;  ws += (size_t)VV * HH * sizeof(_Float16);
  _Float16* hinit    = (_Float16*)ws;  ws += (size_t)BB * HH * sizeof(_Float16);

  // weight conversions (8 elems / thread, vectorized)
  cvt_f16_kernel<<<(HH * FF / 8 + 255) / 256, 256, 0, stream>>>(enc_w_ih, encwih16, HH * FF / 8);
  cvt_f16_kernel<<<(HH * HH / 8 + 255) / 256, 256, 0, stream>>>(enc_w_hh, encwhh16, HH * HH / 8);
  cvt_f16_kernel<<<(HH * VV / 8 + 255) / 256, 256, 0, stream>>>(dec_w_ih, decwih16, HH * VV / 8);
  cvt_f16_kernel<<<(HH * HH / 8 + 255) / 256, 256, 0, stream>>>(dec_w_hh, decwhh16, HH * HH / 8);
  cvt_f16_kernel<<<(VV * HH / 8 + 255) / 256, 256, 0, stream>>>(out_w, outw16, VV * HH / 8);

  // gather + input projection: 256x16 tiles, 8 waves/block
  proj_kernel<<<512, 256, 0, stream>>>(batch, features, encwih16, enc_b_ih, xw);

  // sequential encoder scan (one persistent workgroup)
  encoder_kernel<<<1, 1024, 0, stream>>>(encwhh16, xw, enc_b_hh, hinit);

  // sequential decoder scan (one persistent workgroup, 4096 steps)
  decoder_kernel<<<1, 1024, 0, stream>>>(decwih16, decwhh16, outw16,
                                         dec_b_ih, dec_b_hh, out_b, hinit, out);
}
